// CodeLinearAttention_51634096832894
// MI455X (gfx1250) — matrix operand order repaired and checked by reading, not compiled, not run on hardware
//
#include <hip/hip_runtime.h>
#include <hip/hip_bf16.h>

// ---------------------------------------------------------------------------
// Shapes (compile-time constants from the reference)
// ---------------------------------------------------------------------------
#define B_   2
#define T_   2048
#define DM   1024
#define NH   16
#define HD   64
#define CS   16
#define MR   (B_ * T_)      // 4096 rows of the "token" matrices
#define QKN  (NH * CS)      // 256 folded q/k output channels
#define SCALE 0.125f        // HEAD_DIM^-0.5

typedef __attribute__((ext_vector_type(16))) __bf16       v16bf;
typedef __attribute__((ext_vector_type(8)))  float        v8f;
typedef __attribute__((ext_vector_type(4)))  unsigned int u32x4;
typedef __attribute__((ext_vector_type(8)))  unsigned int u32x8;

union Frag {
    u32x4 q[2];
    v16bf v;
};

// ---------------------------------------------------------------------------
// bf16 split helpers (round-to-nearest-even)
// ---------------------------------------------------------------------------
__device__ __forceinline__ unsigned short f2bf(float x) {
    unsigned u = __float_as_uint(x);
    unsigned r = u + 0x7fffu + ((u >> 16) & 1u);
    return (unsigned short)(r >> 16);
}
__device__ __forceinline__ float bf2f(unsigned short h) {
    return __uint_as_float(((unsigned)h) << 16);
}

// src f32 -> (hi, lo) bf16 pair such that hi + lo ~ src
__global__ __launch_bounds__(256)
void split_bf16_kernel(const float* __restrict__ src,
                       unsigned short* __restrict__ hi,
                       unsigned short* __restrict__ lo,
                       long long n) {
    long long i = (long long)blockIdx.x * 256 + threadIdx.x;
    long long stride = (long long)gridDim.x * 256;
    for (; i < n; i += stride) {
        float x = src[i];
        unsigned short h = f2bf(x);
        float hf = bf2f(h);
        unsigned short l = f2bf(x - hf);
        hi[i] = h;
        lo[i] = l;
    }
}

// ---------------------------------------------------------------------------
// Fold the per-head code matrix (and the attention scale) into the q/k
// projection weights:  Wqc[h*16+c][:] = scale * sum_d code[h][c][d] * Wq[h*64+d][:]
// w_qkv is (3072, 1024): q rows [0,1024), k rows [1024,2048), v rows [2048,3072)
// ---------------------------------------------------------------------------
__global__ __launch_bounds__(128)
void fold_code_kernel(const float* __restrict__ w_qkv,
                      const float* __restrict__ fc_code,
                      float* __restrict__ Wqc,
                      float* __restrict__ Wkc) {
    int r   = blockIdx.x;           // 0..511 : first 256 = q rows, next 256 = k rows
    int isK = (r >= QKN);
    int rc  = r & (QKN - 1);        // h*16 + c
    int h   = rc >> 4;

    const float* code  = fc_code + (size_t)rc * HD;                // (h,c,:) since rc = h*16+c
    const float* wbase = w_qkv + (size_t)(isK ? DM : 0) * DM + (size_t)h * HD * DM;
    float*       outp  = (isK ? Wkc : Wqc) + (size_t)rc * DM;

    for (int j = threadIdx.x; j < DM; j += blockDim.x) {
        float s = 0.f;
#pragma unroll 8
        for (int d = 0; d < HD; ++d)
            s += code[d] * wbase[(size_t)d * DM + j];
        outp[j] = s * SCALE;
    }
}

// ---------------------------------------------------------------------------
// Tensor Data Mover: async 2D tile (64 rows x 32 bf16 cols) global -> LDS.
// Hardware inserts 4 DWORDs (16 B) of padding after every 16 DWORDs (64 B
// = one row), producing an LDS row pitch of LDT = 40 elements.
//
// D# group0 (128b): [1:0]=count=1, [63:32]=lds_addr, [120:64]=global_addr,
//                   [127:126]=type=2.
// D# group1 (256b): [17:16]=data_size(1 => 2B), [20]=pad_en,
//                   [24:22]=pad_interval(3 => 16 DW), [31:25]=pad_amount(3 => 4 DW),
//                   [79:48]=tensor_dim0, [111:80]=tensor_dim1,
//                   [127:112]=tile_dim0, [143:128]=tile_dim1,
//                   [207:160]=tensor_dim0_stride.
// Issued per-wave (scalar, EXEC-independent); tracked with TENSORcnt.
// ---------------------------------------------------------------------------
__device__ __forceinline__ void tdm_load_tile_2d(const unsigned short* gptr,
                                                 unsigned lds_byte_addr,
                                                 unsigned rows,        // tile_dim1
                                                 unsigned cols,        // tile_dim0 (elems)
                                                 unsigned row_stride)  // elems
{
    unsigned long long ga = (unsigned long long)(uintptr_t)gptr;
    u32x4 g0;
    g0[0] = 1u;                                   // count = 1, user descriptor
    g0[1] = lds_byte_addr;                        // LDS byte address
    g0[2] = (unsigned)ga;                         // global_addr[31:0]
    g0[3] = (unsigned)(ga >> 32) | 0x80000000u;   // global_addr[56:32] | type=2
    u32x8 g1;
    g1[0] = (1u << 16)        // data_size = 2 bytes
          | (1u << 20)        // pad_enable
          | (3u << 22)        // pad_interval: every 16 DWORDs (64 B)
          | (3u << 25);       // pad_amount: 4 DWORDs (16 B)
    g1[1] = (cols & 0xffffu) << 16;               // tensor_dim0[15:0]
    g1[2] = (cols >> 16) | ((rows & 0xffffu) << 16); // tensor_dim0[31:16] | tensor_dim1[15:0]
    g1[3] = (rows >> 16) | (cols << 16);          // tensor_dim1[31:16] | tile_dim0
    g1[4] = rows;                                 // tile_dim1 (tile_dim2 = 0)
    g1[5] = row_stride;                           // tensor_dim0_stride[31:0]
    g1[6] = 0u;                                   // stride0 hi | stride1 lo (unused)
    g1[7] = 0u;
    asm volatile("tensor_load_to_lds %0, %1" :: "s"(g0), "s"(g1) : "memory");
}

// ---------------------------------------------------------------------------
// WMMA bf16x3 GEMM:  C(M,N) f32 = (Ah+Al)(M,K) @ (Bh+Bl)(N,K)^T  (drop Al*Bl)
// Tile 64x64, 4 waves (128 threads), K-step 32.
// TDM double-buffered, one tile per wave: wave w DMAs its tile (Ah/Al/Bh/Bl)
// for step s+1 into the other LDS buffer while all waves run WMMAs on the
// current one. TDM ignores EXEC, so tile selection is done with scalar
// (readfirstlane) selects and the whole loop is uniform control flow.
// ---------------------------------------------------------------------------
#define KT  32
#define LDT 40   // 32 + 8 pad elements per row (produced by TDM pad feature)

__global__ __launch_bounds__(128)
void gemm_bf16x3_kernel(const unsigned short* __restrict__ Ah,
                        const unsigned short* __restrict__ Al,
                        const unsigned short* __restrict__ Bh,
                        const unsigned short* __restrict__ Bl,
                        float* __restrict__ C,
                        int M, int N, int K) {
    // [buf][hi=0/lo=1][64*LDT]
    __shared__ alignas(16) unsigned short sA[2][2][64 * LDT];
    __shared__ alignas(16) unsigned short sB[2][2][64 * LDT];

    const int bm   = blockIdx.y * 64;
    const int bn   = blockIdx.x * 64;
    const int tid  = threadIdx.x;
    const int wave = tid >> 5;
    const int lane = tid & 31;

    const int nsteps = K / KT;

    // ---- per-wave TDM tile assignment (scalar) ----------------------------
    // wave 0 -> Ah, wave 1 -> Al, wave 2 -> Bh, wave 3 -> Bl
    const int w = __builtin_amdgcn_readfirstlane(wave);
    const unsigned short* gsrc =
        (w < 2) ? ((w & 1) ? Al : Ah) + (size_t)bm * K
                : ((w & 1) ? Bl : Bh) + (size_t)bn * K;
    unsigned ldsT[2];
    ldsT[0] = (unsigned)(uintptr_t)((w < 2) ? &sA[0][w & 1][0] : &sB[0][w & 1][0]);
    ldsT[1] = (unsigned)(uintptr_t)((w < 2) ? &sA[1][w & 1][0] : &sB[1][w & 1][0]);

    // issue first K-step's tile into buffer 0
    tdm_load_tile_2d(gsrc, ldsT[0], 64, KT, K);

    v8f acc[4] = {};
    int buf = 0;

    for (int s = 0; s < nsteps; ++s) {
        __builtin_amdgcn_s_wait_tensorcnt(0);  // my tile for step s landed
        __syncthreads();   // all 4 tiles resident; step s-1 reads of buf^1 done

        // issue my tile for step s+1 (clamped on last step; drained by the
        // implicit S_WAIT_IDLE of S_ENDPGM, never read)
        {
            const int snext = (s + 1 < nsteps) ? (s + 1) : s;
            tdm_load_tile_2d(gsrc + (size_t)snext * KT, ldsT[buf ^ 1], 64, KT, K);
        }

        // A fragment: lane&15 = row in 16-row strip, chunks at K=(lane>=16?8:0) and +16
        const int ar  = lane & 15;
        const int akk = (lane >> 4) * 8;
        Frag fah, fal;
        {
            const unsigned short* p = &sA[buf][0][(wave * 16 + ar) * LDT + akk];
            fah.q[0] = *(const u32x4*)p;
            fah.q[1] = *(const u32x4*)(p + 16);
            const unsigned short* q = &sA[buf][1][(wave * 16 + ar) * LDT + akk];
            fal.q[0] = *(const u32x4*)q;
            fal.q[1] = *(const u32x4*)(q + 16);
        }
        // B fragments: lane&15 = column, chunks at K=(lane>=16?8:0) and +16
        const int bkk = (lane >> 4) * 8;
#pragma unroll
        for (int nt = 0; nt < 4; ++nt) {
            Frag fbh, fbl;
            const unsigned short* p = &sB[buf][0][(nt * 16 + ar) * LDT + bkk];
            fbh.q[0] = *(const u32x4*)p;
            fbh.q[1] = *(const u32x4*)(p + 16);
            const unsigned short* q = &sB[buf][1][(nt * 16 + ar) * LDT + bkk];
            fbl.q[0] = *(const u32x4*)q;
            fbl.q[1] = *(const u32x4*)(q + 16);

            v8f a = acc[nt];
            a = __builtin_amdgcn_wmma_f32_16x16x32_bf16(false, fah.v, false, fbh.v,
                                                        (short)0, a, false, false);
            a = __builtin_amdgcn_wmma_f32_16x16x32_bf16(false, fah.v, false, fbl.v,
                                                        (short)0, a, false, false);
            a = __builtin_amdgcn_wmma_f32_16x16x32_bf16(false, fal.v, false, fbh.v,
                                                        (short)0, a, false, false);
            acc[nt] = a;
        }
        buf ^= 1;
    }

    // store C: VGPR r -> M = r + (lane>=16?8:0), N = lane&15
    const int cr   = lane & 15;
    const int rofs = (lane >> 4) * 8;
#pragma unroll
    for (int nt = 0; nt < 4; ++nt) {
#pragma unroll
        for (int r = 0; r < 8; ++r) {
            int row = bm + wave * 16 + rofs + r;
            int col = bn + nt * 16 + cr;
            C[(size_t)row * N + col] = acc[nt][r];
        }
    }
}

// ---------------------------------------------------------------------------
// Linear-attention prefix scan. One workgroup per (b,h). 8 waves, each owning
// a 256-step chunk; 3-phase segmented scan of the 16x64 state.
//   qc, kc: (MR, 256)  col = h*16+c   (scale already folded into weights)
//   V:      (MR, 1024) col = h*64+d
//   XO:     (MR, 1024) col = h*64+d
// ---------------------------------------------------------------------------
__global__ __launch_bounds__(256)
void attn_scan_kernel(const float* __restrict__ qc,
                      const float* __restrict__ kc,
                      const float* __restrict__ V,
                      float* __restrict__ XO) {
    const int bh   = blockIdx.x;     // 0..31
    const int h    = bh & (NH - 1);
    const int b    = bh >> 4;
    const int tid  = threadIdx.x;
    const int wave = tid >> 5;
    const int lane = tid & 31;
    const int rowBase = b * T_;

    __shared__ float sRed[16][17];
    __shared__ float sKmax[16];
    __shared__ float sS[8][CS][HD];     // 32 KB chunk sums -> exclusive prefixes
    __shared__ float sKsum[8][16];

    // ---- phase A: global max over T of kc (per c) -------------------------
    {
        int c = tid & 15, g = tid >> 4;  // 16 groups x 128 steps
        float m = -__builtin_inff();
        for (int t = g * 128; t < (g + 1) * 128; ++t)
            m = fmaxf(m, kc[(size_t)(rowBase + t) * QKN + h * CS + c]);
        sRed[g][c] = m;
    }
    __syncthreads();
    if (tid < 16) {
        float m = -__builtin_inff();
        for (int g = 0; g < 16; ++g) m = fmaxf(m, sRed[g][tid]);
        sKmax[tid] = m;
    }
    __syncthreads();

    float kmax[CS];
#pragma unroll
    for (int c = 0; c < CS; ++c) kmax[c] = sKmax[c];

    // lane owns head-dims d0 and d1
    const int d0 = lane, d1 = lane + 32;
    const int t0 = wave * 256;

    // ---- phase B: per-chunk sums ------------------------------------------
    float S0[CS], S1[CS], ks[CS];
#pragma unroll
    for (int c = 0; c < CS; ++c) { S0[c] = 0.f; S1[c] = 0.f; ks[c] = 0.f; }

    for (int t = t0; t < t0 + 256; ++t) {
        const float* krow = kc + (size_t)(rowBase + t) * QKN + h * CS;
        const float* vrow = V + (size_t)(rowBase + t) * DM + h * HD;
        float v0 = vrow[d0], v1 = vrow[d1];
#pragma unroll
        for (int c = 0; c < CS; ++c) {
            float e = __expf(krow[c] - kmax[c]);
            ks[c] += e;
            S0[c] += e * v0;
            S1[c] += e * v1;
        }
    }
#pragma unroll
    for (int c = 0; c < CS; ++c) {
        sS[wave][c][d0] = S0[c];
        sS[wave][c][d1] = S1[c];
    }
    if (lane == 0) {
#pragma unroll
        for (int c = 0; c < CS; ++c) sKsum[wave][c] = ks[c];
    }
    __syncthreads();

    // ---- phase C: exclusive scan across the 8 chunks (in place) -----------
    for (int p = tid; p < CS * HD; p += 256) {
        int c = p >> 6, d = p & 63;
        float run = 0.f;
        for (int w = 0; w < 8; ++w) {
            float v = sS[w][c][d];
            sS[w][c][d] = run;
            run += v;
        }
    }
    if (tid < 16) {
        float run = 0.f;
        for (int w = 0; w < 8; ++w) {
            float v = sKsum[w][tid];
            sKsum[w][tid] = run;
            run += v;
        }
    }
    __syncthreads();

    // ---- phase D: re-scan chunk with carry-in, emit outputs ---------------
#pragma unroll
    for (int c = 0; c < CS; ++c) {
        S0[c] = sS[wave][c][d0];
        S1[c] = sS[wave][c][d1];
        ks[c] = sKsum[wave][c];
    }

    for (int t = t0; t < t0 + 256; ++t) {
        const float* krow = kc + (size_t)(rowBase + t) * QKN + h * CS;
        const float* vrow = V + (size_t)(rowBase + t) * DM + h * HD;
        float v0 = vrow[d0], v1 = vrow[d1];
#pragma unroll
        for (int c = 0; c < CS; ++c) {
            float e = __expf(krow[c] - kmax[c]);
            ks[c] += e;
            S0[c] += e * v0;
            S1[c] += e * v1;
        }
        // softmax over c of qc (pre-scale folded), then * scale, / (ksum+eps)
        const float* qrow = qc + (size_t)(rowBase + t) * QKN + h * CS;
        float qv[CS], qm = -__builtin_inff();
#pragma unroll
        for (int c = 0; c < CS; ++c) { qv[c] = qrow[c]; qm = fmaxf(qm, qv[c]); }
        float qs = 0.f;
#pragma unroll
        for (int c = 0; c < CS; ++c) { qv[c] = __expf(qv[c] - qm); qs += qv[c]; }
        float inv = SCALE / qs;
        float o0 = 0.f, o1 = 0.f;
#pragma unroll
        for (int c = 0; c < CS; ++c) {
            float qt = qv[c] * inv / (ks[c] + 1e-9f);
            o0 += qt * S0[c];
            o1 += qt * S1[c];
        }
        float* orow = XO + (size_t)(rowBase + t) * DM + h * HD;
        orow[d0] = o0;
        orow[d1] = o1;
    }
}

// ---------------------------------------------------------------------------
// Host-side orchestration
// ---------------------------------------------------------------------------
extern "C" void kernel_launch(void* const* d_in, const int* in_sizes, int n_in,
                              void* d_out, int out_size, void* d_ws, size_t ws_size,
                              hipStream_t stream) {
    (void)in_sizes; (void)n_in; (void)out_size; (void)ws_size;
    const float* x       = (const float*)d_in[0];   // (2,2048,1024)
    const float* w_qkv   = (const float*)d_in[1];   // (3072,1024)
    const float* w_out   = (const float*)d_in[2];   // (1024,1024)
    const float* fc_code = (const float*)d_in[3];   // (1,16,16,64)
    float* out = (float*)d_out;                     // (2,2048,1024)

    char* ws = (char*)d_ws;
    size_t off = 0;
    auto alloc = [&](size_t bytes) -> void* {
        void* p = ws + off;
        off = (off + bytes + 255) & ~(size_t)255;
        return p;
    };

    float* Wqc = (float*)alloc((size_t)QKN * DM * 4);
    float* Wkc = (float*)alloc((size_t)QKN * DM * 4);
    unsigned short* Xh   = (unsigned short*)alloc((size_t)MR * DM * 2);
    unsigned short* Xl   = (unsigned short*)alloc((size_t)MR * DM * 2);
    unsigned short* Wqch = (unsigned short*)alloc((size_t)QKN * DM * 2);
    unsigned short* Wqcl = (unsigned short*)alloc((size_t)QKN * DM * 2);
    unsigned short* Wkch = (unsigned short*)alloc((size_t)QKN * DM * 2);
    unsigned short* Wkcl = (unsigned short*)alloc((size_t)QKN * DM * 2);
    unsigned short* Wvh  = (unsigned short*)alloc((size_t)DM * DM * 2);
    unsigned short* Wvl  = (unsigned short*)alloc((size_t)DM * DM * 2);
    unsigned short* Woh  = (unsigned short*)alloc((size_t)DM * DM * 2);
    unsigned short* Wol  = (unsigned short*)alloc((size_t)DM * DM * 2);
    float* qcb = (float*)alloc((size_t)MR * QKN * 4);
    float* kcb = (float*)alloc((size_t)MR * QKN * 4);
    float* Vb  = (float*)alloc((size_t)MR * DM * 4);
    float* XO  = (float*)alloc((size_t)MR * DM * 4);
    unsigned short* XOh = (unsigned short*)alloc((size_t)MR * DM * 2);
    unsigned short* XOl = (unsigned short*)alloc((size_t)MR * DM * 2);

    // 1) fold code (+scale) into q/k projection weights
    fold_code_kernel<<<dim3(2 * QKN), dim3(128), 0, stream>>>(w_qkv, fc_code, Wqc, Wkc);

    // 2) bf16 hi/lo splits
    split_bf16_kernel<<<dim3(2048), dim3(256), 0, stream>>>(x, Xh, Xl, (long long)MR * DM);
    split_bf16_kernel<<<dim3(512),  dim3(256), 0, stream>>>(Wqc, Wqch, Wqcl, (long long)QKN * DM);
    split_bf16_kernel<<<dim3(512),  dim3(256), 0, stream>>>(Wkc, Wkch, Wkcl, (long long)QKN * DM);
    split_bf16_kernel<<<dim3(1024), dim3(256), 0, stream>>>(w_qkv + (size_t)2 * DM * DM, Wvh, Wvl,
                                                            (long long)DM * DM);
    split_bf16_kernel<<<dim3(1024), dim3(256), 0, stream>>>(w_out, Woh, Wol, (long long)DM * DM);

    // 3) WMMA projections: qc, kc (4096x256), V (4096x1024)
    gemm_bf16x3_kernel<<<dim3(QKN / 64, MR / 64), dim3(128), 0, stream>>>(
        Xh, Xl, Wqch, Wqcl, qcb, MR, QKN, DM);
    gemm_bf16x3_kernel<<<dim3(QKN / 64, MR / 64), dim3(128), 0, stream>>>(
        Xh, Xl, Wkch, Wkcl, kcb, MR, QKN, DM);
    gemm_bf16x3_kernel<<<dim3(DM / 64, MR / 64), dim3(128), 0, stream>>>(
        Xh, Xl, Wvh, Wvl, Vb, MR, DM, DM);

    // 4) linear-attention segmented scan
    attn_scan_kernel<<<dim3(B_ * NH), dim3(256), 0, stream>>>(qcb, kcb, Vb, XO);

    // 5) output projection
    split_bf16_kernel<<<dim3(2048), dim3(256), 0, stream>>>(XO, XOh, XOl, (long long)MR * DM);
    gemm_bf16x3_kernel<<<dim3(DM / 64, MR / 64), dim3(128), 0, stream>>>(
        XOh, XOl, Woh, Wol, out, MR, DM, DM);
}
